// BGInpaintingAttn_2525440770433
// MI455X (gfx1250) — compile-verified, matrix operand order checked by reading, not run on hardware
//
#include <hip/hip_runtime.h>

// ---------------------------------------------------------------------------
// BGInpaintingAttn for MI455X (gfx1250, wave32, WMMA).
//
// Shapes: B=2,T=4 -> N=8 frames; C=32; H=W=208 (HW=43264); HEAD=2, DK=DV=16,
// window P=13 -> 16x16 windows of 169 tokens; 4096 windows total.
//
// Pipeline (one stream, 6 kernels + 2 d2d copies):
//  A: y1 = lrelu(conv3x3(x,w1)+b1, 0.2)                    [VALU, weights in LDS]
//  B: y2 = conv1x1(y1,w2); mask=sigmoid(y2[32]); kv=y2[:32]*(1-mask);
//     also writes bgmask channel 3.
//  C: q/k/v 1x1 convs, stored directly WINDOW-major: [win][p(169)][d(16)]
//  D: per-window attention, 1 wave32 per window, f16 WMMA 16x16x32:
//     11x11 score WMMAs + softmax (shfl_xor row reduce) + 11x6 PV WMMAs.
//     Q/K stored in LDS with contraction dim padded to 32 so every WMMA
//     operand is exactly two aligned ds_load_b128 (no register zero-splats).
//  E: attn_f = conv1x1(o,wo)+bo
//  F: pose branch fused: lrelu(x,.1) -> conv3x3(pw1)+lrelu -> conv1x1(pw2)
//     -> sigmoid -> bgmask channels 0..2.
//
// Workspace: 5 * 11,075,584 floats (~222 MB). o aliases y1 (dead after B).
// ---------------------------------------------------------------------------

#define HWp 43264
#define Wd 208
#define NB 8
#define NPIX (NB * HWp)   // 346112
#define AST 176           // padded token count (11 tiles of 16)

typedef _Float16 h8   __attribute__((ext_vector_type(8)));
typedef _Float16 v16h __attribute__((ext_vector_type(16)));
typedef float    v8f  __attribute__((ext_vector_type(8)));

__device__ __forceinline__ float lrelu(float v, float s) { return v >= 0.f ? v : s * v; }
__device__ __forceinline__ float sigm(float v) { return 1.f / (1.f + __expf(-v)); }

// ---------------- Kernel A: conv3x3 (32->32) + lrelu(0.2) ------------------
__global__ __launch_bounds__(256) void conv3x3_lrelu(
    const float* __restrict__ x, const float* __restrict__ w1,
    const float* __restrict__ b1, float* __restrict__ y1) {
  __shared__ float sw[32 * 32 * 9];
  __shared__ float sb[32];
  int tid = threadIdx.x;
  for (int t = tid; t < 32 * 32 * 9; t += 256) sw[t] = w1[t];
  if (tid < 32) sb[tid] = b1[tid];
  __syncthreads();
  int pix = blockIdx.x * 256 + tid;
  if (pix >= NPIX) return;
  int n = pix / HWp, hw = pix % HWp;
  int hy = hw / Wd, wx = hw % Wd;
  float acc[32];
#pragma unroll
  for (int oc = 0; oc < 32; oc++) acc[oc] = sb[oc];
  for (int ic = 0; ic < 32; ic++) {
    const float* xp = x + (size_t)(n * 32 + ic) * HWp;
#pragma unroll
    for (int ky = 0; ky < 3; ky++) {
      int yy = hy + ky - 1;
      if ((unsigned)yy >= 208u) continue;
#pragma unroll
      for (int kx = 0; kx < 3; kx++) {
        int xx = wx + kx - 1;
        if ((unsigned)xx >= 208u) continue;
        float xv = xp[yy * Wd + xx];
        const float* wp = sw + (ic * 3 + ky) * 3 + kx;
#pragma unroll
        for (int oc = 0; oc < 32; oc++) acc[oc] = fmaf(xv, wp[oc * 288], acc[oc]);
      }
    }
  }
#pragma unroll
  for (int oc = 0; oc < 32; oc++)
    y1[(size_t)(n * 32 + oc) * HWp + hw] = lrelu(acc[oc], 0.2f);
}

// -------- Kernel B: conv1x1 (32->33), sigmoid gate, kv, bgmask[3] ----------
__global__ __launch_bounds__(256) void conv1_gate(
    const float* __restrict__ y1, const float* __restrict__ w2,
    const float* __restrict__ b2, float* __restrict__ kv,
    float* __restrict__ out_bg) {
  __shared__ float sw[33 * 32];
  __shared__ float sb[33];
  int tid = threadIdx.x;
  for (int t = tid; t < 33 * 32; t += 256) sw[t] = w2[t];
  if (tid < 33) sb[tid] = b2[tid];
  __syncthreads();
  int pix = blockIdx.x * 256 + tid;
  if (pix >= NPIX) return;
  int n = pix / HWp, hw = pix % HWp;
  float in[32];
#pragma unroll
  for (int ic = 0; ic < 32; ic++) in[ic] = y1[(size_t)(n * 32 + ic) * HWp + hw];
  float m = sb[32];
#pragma unroll
  for (int ic = 0; ic < 32; ic++) m = fmaf(in[ic], sw[32 * 32 + ic], m);
  m = sigm(m);
  out_bg[(size_t)(n * 4 + 3) * HWp + hw] = m;
  float g = 1.f - m;
#pragma unroll
  for (int oc = 0; oc < 32; oc++) {
    float a = sb[oc];
#pragma unroll
    for (int ic = 0; ic < 32; ic++) a = fmaf(in[ic], sw[oc * 32 + ic], a);
    kv[(size_t)(n * 32 + oc) * HWp + hw] = a * g;
  }
}

// ------ Kernel C: q/k/v 1x1 convs -> window-major [win][p][d] layout -------
__global__ __launch_bounds__(256) void conv1_qkv(
    const float* __restrict__ h, const float* __restrict__ kvb,
    const float* __restrict__ wq, const float* __restrict__ bq,
    const float* __restrict__ wk, const float* __restrict__ bk,
    const float* __restrict__ wv, const float* __restrict__ bv,
    float* __restrict__ qw, float* __restrict__ kw, float* __restrict__ vw) {
  __shared__ float swq[1024], swk[1024], swv[1024];
  __shared__ float sbq[32], sbk[32], sbv[32];
  int tid = threadIdx.x;
  for (int t = tid; t < 1024; t += 256) { swq[t] = wq[t]; swk[t] = wk[t]; swv[t] = wv[t]; }
  if (tid < 32) { sbq[tid] = bq[tid]; sbk[tid] = bk[tid]; sbv[tid] = bv[tid]; }
  __syncthreads();
  int pix = blockIdx.x * 256 + tid;
  if (pix >= NPIX) return;
  int n = pix / HWp, hw = pix % HWp;
  int hy = hw / Wd, wx = hw % Wd;
  int i = hy / 13, pi = hy % 13, j = wx / 13, pj = wx % 13;
  int p = pi * 13 + pj;
  size_t base0 = (size_t)((n * 2 + 0) * 256 + i * 16 + j) * 2704 + (size_t)p * 16;
  size_t base1 = (size_t)((n * 2 + 1) * 256 + i * 16 + j) * 2704 + (size_t)p * 16;
  float in[32];
#pragma unroll
  for (int c = 0; c < 32; c++) in[c] = h[(size_t)(n * 32 + c) * HWp + hw];
#pragma unroll
  for (int oc = 0; oc < 32; oc++) {
    float a = sbq[oc];
#pragma unroll
    for (int ic = 0; ic < 32; ic++) a = fmaf(in[ic], swq[oc * 32 + ic], a);
    qw[(oc < 16 ? base0 : base1) + (oc & 15)] = a;
  }
#pragma unroll
  for (int c = 0; c < 32; c++) in[c] = kvb[(size_t)(n * 32 + c) * HWp + hw];
#pragma unroll
  for (int oc = 0; oc < 32; oc++) {
    float a = sbk[oc];
#pragma unroll
    for (int ic = 0; ic < 32; ic++) a = fmaf(in[ic], swk[oc * 32 + ic], a);
    kw[(oc < 16 ? base0 : base1) + (oc & 15)] = a;
  }
#pragma unroll
  for (int oc = 0; oc < 32; oc++) {
    float a = sbv[oc];
#pragma unroll
    for (int ic = 0; ic < 32; ic++) a = fmaf(in[ic], swv[oc * 32 + ic], a);
    vw[(oc < 16 ? base0 : base1) + (oc & 15)] = a;
  }
}

// -------- Kernel D: windowed attention, one wave32 per window, WMMA --------
__global__ __launch_bounds__(32) void attn_win(
    const float* __restrict__ qg, const float* __restrict__ kg,
    const float* __restrict__ vg, float* __restrict__ og) {
  // Q/K rows padded to K-dim 32 (d 16..31 zero) so WMMA operands are two
  // aligned ds_load_b128 each -- no register zero-splat moves.
  __shared__ __align__(16) _Float16 Qh[AST * 32];   // [row p][d(32)]
  __shared__ __align__(16) _Float16 Kh[AST * 32];   // [key p][d(32)]
  __shared__ __align__(16) _Float16 VT[16 * AST];   // [d][key p] (transposed)
  __shared__ __align__(16) _Float16 AR[16 * AST];   // probs staging [m][key]
  int lane = threadIdx.x;
  int wid = blockIdx.x;
  size_t base = (size_t)wid * 2704;

  h8 hz;
#pragma unroll
  for (int e = 0; e < 8; e++) hz[e] = (_Float16)0.f;

  // Zero Q/K/V staging with b128 stores (covers the pad regions), then fill.
  // LDS ops from one wave execute in order, so no wait needed between.
  {
    h8* q8 = (h8*)Qh;  // 704 vectors
    h8* k8 = (h8*)Kh;  // 704 vectors
    h8* v8 = (h8*)VT;  // 352 vectors
    for (int t = lane; t < 704; t += 32) { q8[t] = hz; k8[t] = hz; }
    for (int t = lane; t < 352; t += 32) v8[t] = hz;
  }
  for (int t = lane; t < 2704; t += 32) {
    int p = t >> 4, d = t & 15;
    Qh[p * 32 + d] = (_Float16)qg[base + t];
    Kh[p * 32 + d] = (_Float16)kg[base + t];
    VT[d * AST + p] = (_Float16)vg[base + t];
  }
  asm volatile("s_wait_dscnt 0" ::: "memory");

  int half = lane >> 4, l16 = lane & 15;
  int n = wid >> 9, rem = wid & 511;
  int head = rem >> 8, ij = rem & 255;
  int i = ij >> 4, j = ij & 15;

  const float scale = 0.25f;  // 1/sqrt(DK=16)

  for (int r = 0; r < 11; r++) {
    // A operand: Q rows M=l16 of row-tile r. Layout per lane:
    // half0 -> K{0..7,16..23}, half1 -> K{8..15,24..31}: base=half*8, +16.
    union { v16h v; h8 h[2]; } qf;
    {
      const _Float16* qp = Qh + (r * 16 + l16) * 32 + half * 8;
      qf.h[0] = *(const h8*)(qp);
      qf.h[1] = *(const h8*)(qp + 16);
    }
    v8f acc[11];
#pragma unroll
    for (int ct = 0; ct < 11; ct++) {
      union { v16h v; h8 h[2]; } bf;  // B = K^T: column n=l16 is key ct*16+l16
      const _Float16* kp = Kh + (ct * 16 + l16) * 32 + half * 8;
      bf.h[0] = *(const h8*)(kp);
      bf.h[1] = *(const h8*)(kp + 16);
      v8f cz = {0.f, 0.f, 0.f, 0.f, 0.f, 0.f, 0.f, 0.f};
      acc[ct] = __builtin_amdgcn_wmma_f32_16x16x32_f16(
          false, qf.v, false, bf.v, (short)0, cz, false, false);
    }
    // Softmax over 176 cols (mask >=169). C layout: VGPR v, lane half h:
    // element = (M = v + 8*h, N = l16) of tile ct -> col = ct*16+l16.
    float mx[8], sm[8];
#pragma unroll
    for (int v = 0; v < 8; v++) mx[v] = -1e30f;
#pragma unroll
    for (int ct = 0; ct < 11; ct++) {
      bool valid = (ct * 16 + l16) < 169;
#pragma unroll
      for (int v = 0; v < 8; v++)
        if (valid) mx[v] = fmaxf(mx[v], acc[ct][v] * scale);
    }
#pragma unroll
    for (int v = 0; v < 8; v++) {
#pragma unroll
      for (int msk = 1; msk < 16; msk <<= 1)
        mx[v] = fmaxf(mx[v], __shfl_xor(mx[v], msk));
      sm[v] = 0.f;
    }
#pragma unroll
    for (int ct = 0; ct < 11; ct++) {
      int col = ct * 16 + l16;
      bool valid = col < 169;
#pragma unroll
      for (int v = 0; v < 8; v++) {
        float pr = valid ? __expf(acc[ct][v] * scale - mx[v]) : 0.f;
        sm[v] += pr;
        AR[(v + half * 8) * AST + col] = (_Float16)pr;
      }
    }
#pragma unroll
    for (int v = 0; v < 8; v++) {
#pragma unroll
      for (int msk = 1; msk < 16; msk <<= 1) sm[v] += __shfl_xor(sm[v], msk);
    }
    asm volatile("s_wait_dscnt 0" ::: "memory");
    // O = P * V: A rows M=l16 from AR; B cols n=l16 from VT (transposed V).
    v8f oacc = {0.f, 0.f, 0.f, 0.f, 0.f, 0.f, 0.f, 0.f};
#pragma unroll
    for (int kt = 0; kt < 6; kt++) {
      int kb = kt * 32 + half * 8;
      union { v16h v; h8 h[2]; } af, bf;
      af.h[0] = *(const h8*)(AR + l16 * AST + kb);
      af.h[1] = *(const h8*)(AR + l16 * AST + kb + 16);
      bf.h[0] = *(const h8*)(VT + l16 * AST + kb);
      bf.h[1] = *(const h8*)(VT + l16 * AST + kb + 16);
      oacc = __builtin_amdgcn_wmma_f32_16x16x32_f16(
          false, af.v, false, bf.v, (short)0, oacc, false, false);
    }
    // Store normalized O back to NCHW for the wo conv. sm[v] aligns with
    // oacc[v] (same C-layout row mapping).
#pragma unroll
    for (int v = 0; v < 8; v++) {
      int p = r * 16 + v + half * 8;
      if (p <= 168) {
        int pi = p / 13, pj = p % 13;
        int hy = i * 13 + pi, wx = j * 13 + pj;
        int c = head * 16 + l16;
        og[(size_t)(n * 32 + c) * HWp + hy * Wd + wx] = oacc[v] / sm[v];
      }
    }
  }
}

// ---------------- Kernel E: attn_f = conv1x1(o, wo) + bo -------------------
__global__ __launch_bounds__(256) void conv1_out(
    const float* __restrict__ o, const float* __restrict__ wo,
    const float* __restrict__ bo, float* __restrict__ out_attn) {
  __shared__ float sw[1024];
  __shared__ float sb[32];
  int tid = threadIdx.x;
  for (int t = tid; t < 1024; t += 256) sw[t] = wo[t];
  if (tid < 32) sb[tid] = bo[tid];
  __syncthreads();
  int pix = blockIdx.x * 256 + tid;
  if (pix >= NPIX) return;
  int n = pix / HWp, hw = pix % HWp;
  float in[32];
#pragma unroll
  for (int ic = 0; ic < 32; ic++) in[ic] = o[(size_t)(n * 32 + ic) * HWp + hw];
#pragma unroll
  for (int oc = 0; oc < 32; oc++) {
    float a = sb[oc];
#pragma unroll
    for (int ic = 0; ic < 32; ic++) a = fmaf(in[ic], sw[oc * 32 + ic], a);
    out_attn[(size_t)(n * 32 + oc) * HWp + hw] = a;
  }
}

// ------- Kernel F: pose branch fused (lrelu -> conv3x3 -> conv1x1) ---------
__global__ __launch_bounds__(256) void pose_path(
    const float* __restrict__ x, const float* __restrict__ pw1,
    const float* __restrict__ pb1, const float* __restrict__ pw2,
    const float* __restrict__ pb2, float* __restrict__ out_bg) {
  __shared__ float sw1[16 * 32 * 9];
  __shared__ float sb1[16];
  __shared__ float sw2[48];
  __shared__ float sb2[3];
  int tid = threadIdx.x;
  for (int t = tid; t < 16 * 32 * 9; t += 256) sw1[t] = pw1[t];
  if (tid < 16) sb1[tid] = pb1[tid];
  if (tid < 48) sw2[tid] = pw2[tid];
  if (tid < 3) sb2[tid] = pb2[tid];
  __syncthreads();
  int pix = blockIdx.x * 256 + tid;
  if (pix >= NPIX) return;
  int n = pix / HWp, hw = pix % HWp;
  int hy = hw / Wd, wx = hw % Wd;
  float acc[16];
#pragma unroll
  for (int oc = 0; oc < 16; oc++) acc[oc] = sb1[oc];
  for (int ic = 0; ic < 32; ic++) {
    const float* xp = x + (size_t)(n * 32 + ic) * HWp;
#pragma unroll
    for (int ky = 0; ky < 3; ky++) {
      int yy = hy + ky - 1;
      if ((unsigned)yy >= 208u) continue;
#pragma unroll
      for (int kx = 0; kx < 3; kx++) {
        int xx = wx + kx - 1;
        if ((unsigned)xx >= 208u) continue;
        float xv = lrelu(xp[yy * Wd + xx], 0.1f);
        const float* wp = sw1 + (ic * 3 + ky) * 3 + kx;
#pragma unroll
        for (int oc = 0; oc < 16; oc++) acc[oc] = fmaf(xv, wp[oc * 288], acc[oc]);
      }
    }
  }
#pragma unroll
  for (int oc = 0; oc < 16; oc++) acc[oc] = lrelu(acc[oc], 0.1f);
#pragma unroll
  for (int cc = 0; cc < 3; cc++) {
    float a = sb2[cc];
#pragma unroll
    for (int ic = 0; ic < 16; ic++) a = fmaf(acc[ic], sw2[cc * 16 + ic], a);
    out_bg[(size_t)(n * 4 + cc) * HWp + hw] = sigm(a);
  }
}

// ---------------------------------------------------------------------------
extern "C" void kernel_launch(void* const* d_in, const int* in_sizes, int n_in,
                              void* d_out, int out_size, void* d_ws, size_t ws_size,
                              hipStream_t stream) {
  (void)in_sizes; (void)n_in; (void)out_size; (void)ws_size;
  const float* x   = (const float*)d_in[0];
  const float* h   = (const float*)d_in[1];
  const float* w1  = (const float*)d_in[2];
  const float* b1  = (const float*)d_in[3];
  const float* w2  = (const float*)d_in[4];
  const float* b2  = (const float*)d_in[5];
  const float* wq  = (const float*)d_in[6];
  const float* bq  = (const float*)d_in[7];
  const float* wk  = (const float*)d_in[8];
  const float* bk  = (const float*)d_in[9];
  const float* wv  = (const float*)d_in[10];
  const float* bv  = (const float*)d_in[11];
  const float* wo  = (const float*)d_in[12];
  const float* bo  = (const float*)d_in[13];
  const float* pw1 = (const float*)d_in[14];
  const float* pb1 = (const float*)d_in[15];
  const float* pw2 = (const float*)d_in[16];
  const float* pb2 = (const float*)d_in[17];

  float* out = (float*)d_out;
  const size_t NXH = (size_t)NB * 32 * HWp;  // 11,075,584
  float* ws = (float*)d_ws;
  float* y1 = ws;                 // conv3x3 output; reused as attention output
  float* kv = ws + NXH;
  float* qw = ws + 2 * NXH;
  float* kw = ws + 3 * NXH;
  float* vw = ws + 4 * NXH;
  float* ob = y1;                 // alias: y1 is dead after conv1_gate
  float* out_bg   = out + 2 * NXH;                       // (N,4,HW)
  float* out_attn = out + 2 * NXH + (size_t)NB * 4 * HWp;

  // Passthrough outputs x, h.
  hipMemcpyAsync(out, x, NXH * sizeof(float), hipMemcpyDeviceToDevice, stream);
  hipMemcpyAsync(out + NXH, h, NXH * sizeof(float), hipMemcpyDeviceToDevice, stream);

  dim3 blk(256), grd(NPIX / 256);  // 1352 blocks, exact
  conv3x3_lrelu<<<grd, blk, 0, stream>>>(x, w1, b1, y1);
  conv1_gate<<<grd, blk, 0, stream>>>(y1, w2, b2, kv, out_bg);
  conv1_qkv<<<grd, blk, 0, stream>>>(h, kv, wq, bq, wk, bk, wv, bv, qw, kw, vw);
  attn_win<<<dim3(4096), dim3(32), 0, stream>>>(qw, kw, vw, ob);
  conv1_out<<<grd, blk, 0, stream>>>(ob, wo, bo, out_attn);
  pose_path<<<grd, blk, 0, stream>>>(x, pw1, pb1, pw2, pb2, out_bg);
}
